// PerPosScaleMatryoshkaTXC_50036368999040
// MI455X (gfx1250) — matrix operand order repaired and checked by reading, not compiled, not run on hardware
//
#include <hip/hip_runtime.h>
#include <hip/hip_bf16.h>
#include <stdint.h>

// ---------------------------------------------------------------------------
// PerPosScaleMatryoshkaTXC  (MI455X / gfx1250, wave32)
// Pipeline: invperm -> enc GEMM (split-bf16 WMMA 16x16x32) -> topK ->
//           zero z -> scatter -> sparse decode + loss partials -> loss reduce
//
// Encoder GEMM uses x = hi + lo bf16 splitting:  x*w ~= hh + hl + lh, each an
// exact-product bf16 WMMA with f32 accumulate; residual ~2^-17 relative,
// far below the ~0.007 top-k order-statistic gap -> ranking-safe.
// ---------------------------------------------------------------------------

typedef float   v8f  __attribute__((ext_vector_type(8)));
typedef __bf16  v16bf __attribute__((ext_vector_type(16)));
typedef __bf16  v4bf  __attribute__((ext_vector_type(4)));

#define D_IN   768
#define D_SAE  16384
#define T_POS  4
#define KTOP   64
#define BATCH  1024
#define KTOT   (T_POS * D_IN)       // 3072
#define PFX0   8192

union FragBF { v16bf v; v4bf q[4]; };

__device__ __forceinline__ void split_bf16(float x, __bf16& h, __bf16& l) {
    h = (__bf16)x;
    l = (__bf16)(x - (float)h);
}

// ---------------------------------------------------------------------------
// Kernel 1: inverse permutation  inv[t][perm[t][p]] = p
// ---------------------------------------------------------------------------
__global__ __launch_bounds__(256) void invperm_kernel(const int* __restrict__ perm,
                                                      int* __restrict__ inv) {
    int i = blockIdx.x * 256 + threadIdx.x;       // i = t*16384 + p
    int t = i >> 14;
    int p = i & (D_SAE - 1);
    inv[t * D_SAE + perm[i]] = p;
}

// ---------------------------------------------------------------------------
// Kernel 2: encoder GEMM  pre[b][s] = sum_k X[b][k] * W[k][s] + b_enc[s]
// X: (1024 x 3072), W: (3072 x 16384), both f32 row-major in global.
// Block tile 64(M) x 256(N), BK=32; 8 waves, wave tile 32(M) x 64(N).
// LDS staged as split bf16 hi/lo in fragment-native order:
//   A:  [row][k]  k-contiguous,  B: transposed [col][k] k-contiguous.
// Per chunk per wave: 2(m) x 4(n) x 3(hh,hl,lh) = 24 v_wmma_f32_16x16x32_bf16.
// ---------------------------------------------------------------------------
__global__ __launch_bounds__(256) void enc_gemm_kernel(const float* __restrict__ X,
                                                       const float* __restrict__ W,
                                                       const float* __restrict__ benc,
                                                       float* __restrict__ pre) {
    __shared__ __bf16 Ahi[64][36];      // stride 36 bf16 = 72B: 8B-aligned rows,
    __shared__ __bf16 Alo[64][36];      // odd bank stride -> conflict-light
    __shared__ __bf16 Bhi[256][36];
    __shared__ __bf16 Blo[256][36];

    const int tid  = threadIdx.x;
    const int lane = tid & 31;
    const int wave = tid >> 5;
    const int wm   = wave & 1;          // M strip (32 rows)
    const int wn   = wave >> 1;         // N strip (64 cols)
    const int half = lane >> 4;         // fragment K-half select
    const int ml   = lane & 15;         // M (A) / N (B) within fragment

    const int blockM = blockIdx.y * 64;
    const int blockN = blockIdx.x * 256;

    v8f acc[2][4] = {};                 // 2 M sub-tiles x 4 N sub-tiles

    for (int k0 = 0; k0 < KTOT; k0 += 32) {
        // ---- stage A: 64 x 32 f32 -> bf16 hi/lo (2 float4 per thread) ----
#pragma unroll
        for (int i = 0; i < 2; i++) {
            int f  = tid + 256 * i;
            int r  = f >> 3;
            int kq = (f & 7) << 2;
            float4 a4 = *reinterpret_cast<const float4*>(
                X + (size_t)(blockM + r) * KTOT + k0 + kq);
            float vv[4] = {a4.x, a4.y, a4.z, a4.w};
#pragma unroll
            for (int j = 0; j < 4; j++) {
                __bf16 h, l;
                split_bf16(vv[j], h, l);
                Ahi[r][kq + j] = h;
                Alo[r][kq + j] = l;
            }
        }
        // ---- stage B: 32 x 256 f32 -> bf16 hi/lo, transposed (8 float4) ----
#pragma unroll
        for (int i = 0; i < 8; i++) {
            int f  = tid + 256 * i;
            int r  = f >> 6;            // k row 0..31
            int cq = (f & 63) << 2;     // col 0..252
            float4 b4 = *reinterpret_cast<const float4*>(
                W + (size_t)(k0 + r) * D_SAE + blockN + cq);
            float vv[4] = {b4.x, b4.y, b4.z, b4.w};
#pragma unroll
            for (int j = 0; j < 4; j++) {
                __bf16 h, l;
                split_bf16(vv[j], h, l);
                Bhi[cq + j][r] = h;
                Blo[cq + j][r] = l;
            }
        }
        __syncthreads();

        // ---- fragments ----
        // A frag (16x32 bf16): lanes 0-15: K 0..7 (v0-3), 16..23 (v4-7);
        //                      lanes 16-31: K 8..15, 24..31  -> offsets 8h, 16+8h
        FragBF ah[2], al[2];
#pragma unroll
        for (int mi = 0; mi < 2; mi++) {
            const int m = wm * 32 + mi * 16 + ml;
            const v4bf* ph0 = reinterpret_cast<const v4bf*>(&Ahi[m][8 * half]);
            const v4bf* ph1 = reinterpret_cast<const v4bf*>(&Ahi[m][16 + 8 * half]);
            ah[mi].q[0] = ph0[0]; ah[mi].q[1] = ph0[1];
            ah[mi].q[2] = ph1[0]; ah[mi].q[3] = ph1[1];
            const v4bf* pl0 = reinterpret_cast<const v4bf*>(&Alo[m][8 * half]);
            const v4bf* pl1 = reinterpret_cast<const v4bf*>(&Alo[m][16 + 8 * half]);
            al[mi].q[0] = pl0[0]; al[mi].q[1] = pl0[1];
            al[mi].q[2] = pl1[0]; al[mi].q[3] = pl1[1];
        }

        // B frag (32x16 bf16): same K order as A -> offsets 8h, 16+8h
#pragma unroll
        for (int cn = 0; cn < 4; cn++) {
            const int col = wn * 64 + cn * 16 + ml;
            FragBF bh, bl;
            const v4bf* pbh = reinterpret_cast<const v4bf*>(&Bhi[col][8 * half]);
            const v4bf* pbh1 = reinterpret_cast<const v4bf*>(&Bhi[col][16 + 8 * half]);
            bh.q[0] = pbh[0]; bh.q[1] = pbh[1]; bh.q[2] = pbh1[0]; bh.q[3] = pbh1[1];
            const v4bf* pbl = reinterpret_cast<const v4bf*>(&Blo[col][8 * half]);
            const v4bf* pbl1 = reinterpret_cast<const v4bf*>(&Blo[col][16 + 8 * half]);
            bl.q[0] = pbl[0]; bl.q[1] = pbl[1]; bl.q[2] = pbl1[0]; bl.q[3] = pbl1[1];

#pragma unroll
            for (int mi = 0; mi < 2; mi++) {
                acc[mi][cn] = __builtin_amdgcn_wmma_f32_16x16x32_bf16(
                    false, ah[mi].v, false, bh.v, (short)0, acc[mi][cn], false, false);
                acc[mi][cn] = __builtin_amdgcn_wmma_f32_16x16x32_bf16(
                    false, ah[mi].v, false, bl.v, (short)0, acc[mi][cn], false, false);
                acc[mi][cn] = __builtin_amdgcn_wmma_f32_16x16x32_bf16(
                    false, al[mi].v, false, bh.v, (short)0, acc[mi][cn], false, false);
            }
        }
        __syncthreads();
    }

    // epilogue: C layout — VGPR r: lanes 0-15 -> M=r, lanes 16-31 -> M=r+8
#pragma unroll
    for (int cn = 0; cn < 4; cn++) {
        const int gcol = blockN + wn * 64 + cn * 16 + ml;
        const float bias = benc[gcol];
#pragma unroll
        for (int mi = 0; mi < 2; mi++) {
#pragma unroll
            for (int r = 0; r < 8; r++) {
                const int grow = blockM + wm * 32 + mi * 16 + r + 8 * half;
                pre[(size_t)grow * D_SAE + gcol] = acc[mi][cn][r] + bias;
            }
        }
    }
}

// ---------------------------------------------------------------------------
// Kernel 3: exact top-64 per row via strictly-decreasing-key argmax.
// Key = (orderedFloat(val) << 32) | ~idx  -> unique keys, ties to lowest idx.
// ---------------------------------------------------------------------------
__device__ __forceinline__ unsigned int float_key(float f) {
    unsigned int u = __float_as_uint(f);
    return (u & 0x80000000u) ? ~u : (u | 0x80000000u);
}

__global__ __launch_bounds__(256) void topk_kernel(const float* __restrict__ pre,
                                                   int* __restrict__ idxs,
                                                   float* __restrict__ vals) {
    const int tid = threadIdx.x;
    const size_t base = (size_t)blockIdx.x * D_SAE;

    unsigned int uk[64];
#pragma unroll
    for (int jj = 0; jj < 64; jj++)
        uk[jj] = float_key(pre[base + tid + jj * 256]);

    __shared__ unsigned long long red[256];
    unsigned long long lastKey = ~0ull;

    for (int it = 0; it < KTOP; it++) {
        unsigned long long best = 0ull;
#pragma unroll
        for (int jj = 0; jj < 64; jj++) {
            const unsigned int j = (unsigned int)(tid + jj * 256);
            const unsigned long long key =
                ((unsigned long long)uk[jj] << 32) | (unsigned int)~j;
            if (key < lastKey && key > best) best = key;
        }
        red[tid] = best;
        __syncthreads();
        for (int s = 128; s > 0; s >>= 1) {
            if (tid < s && red[tid + s] > red[tid]) red[tid] = red[tid + s];
            __syncthreads();
        }
        const unsigned long long k = red[0];
        if (tid == 0) {
            const unsigned int hi = (unsigned int)(k >> 32);
            const unsigned int bits =
                (hi & 0x80000000u) ? (hi ^ 0x80000000u) : ~hi;
            const float v = __uint_as_float(bits);
            const int j = (int)~(unsigned int)(k & 0xFFFFFFFFu);
            idxs[blockIdx.x * KTOP + it] = j;
            vals[blockIdx.x * KTOP + it] = fmaxf(v, 0.0f);   // relu
        }
        lastKey = k;
        __syncthreads();
    }
}

// ---------------------------------------------------------------------------
// Kernel 4: zero z region (the pre scratch becomes z)
// ---------------------------------------------------------------------------
__global__ __launch_bounds__(256) void zero_z_kernel(float* __restrict__ z) {
    const size_t stride = (size_t)gridDim.x * blockDim.x;
    for (size_t i = (size_t)blockIdx.x * blockDim.x + threadIdx.x;
         i < (size_t)BATCH * D_SAE; i += stride)
        z[i] = 0.0f;
}

// ---------------------------------------------------------------------------
// Kernel 5: scatter relu(vals) at idx
// ---------------------------------------------------------------------------
__global__ __launch_bounds__(256) void scatter_kernel(const int* __restrict__ idxs,
                                                      const float* __restrict__ vals,
                                                      float* __restrict__ z) {
    const int i = blockIdx.x * 256 + threadIdx.x;      // < 1024*64
    const int row = i >> 6;
    z[(size_t)row * D_SAE + idxs[i]] = vals[i];
}

// ---------------------------------------------------------------------------
// Kernel 6: sparse decode + per-block loss partials.
// One block per (b, t). Thread handles d, d+256, d+512.
// ---------------------------------------------------------------------------
__global__ __launch_bounds__(256) void decode_kernel(
    const float* __restrict__ x,
    const int* __restrict__ idxs, const float* __restrict__ vals,
    const int* __restrict__ inv,
    const float* __restrict__ Wd0, const float* __restrict__ bd0,
    const float* __restrict__ Wd1, const float* __restrict__ bd1,
    float* __restrict__ xhat, float* __restrict__ bsums) {
    const int b = blockIdx.x, t = blockIdx.y, tid = threadIdx.x;

    __shared__ int   sp[KTOP];
    __shared__ float sv[KTOP];
    __shared__ float sred[256];

    if (tid < KTOP) {
        const int j = idxs[b * KTOP + tid];
        sv[tid] = vals[b * KTOP + tid];
        sp[tid] = inv[t * D_SAE + j];
    }
    __syncthreads();

    float acc0[3] = {0.f, 0.f, 0.f};
    float acc1[3] = {0.f, 0.f, 0.f};

    for (int i = 0; i < KTOP; i++) {
        const float v = sv[i];
        if (v != 0.0f) {
            const int p = sp[i];
            const float* w1 = Wd1 + ((size_t)p * T_POS + t) * D_IN;
#pragma unroll
            for (int k = 0; k < 3; k++)
                acc1[k] = fmaf(v, w1[tid + k * 256], acc1[k]);
            if (p < PFX0) {
                const float* w0 = Wd0 + ((size_t)p * T_POS + t) * D_IN;
#pragma unroll
                for (int k = 0; k < 3; k++)
                    acc0[k] = fmaf(v, w0[tid + k * 256], acc0[k]);
            }
        }
    }

    float lsum = 0.0f;
#pragma unroll
    for (int k = 0; k < 3; k++) {
        const int d = tid + k * 256;
        const size_t off = ((size_t)b * T_POS + t) * D_IN + d;
        const float xv = x[off];
        const float h1 = acc1[k] + bd1[t * D_IN + d];
        xhat[off] = h1;                      // x_hat_full = last decoder
        const float d1 = h1 - xv;
        const float h0 = acc0[k] + bd0[t * D_IN + d];
        const float d0 = h0 - xv;
        lsum += d0 * d0 + d1 * d1;
    }

    sred[tid] = lsum;
    __syncthreads();
    for (int s = 128; s > 0; s >>= 1) {
        if (tid < s) sred[tid] += sred[tid + s];
        __syncthreads();
    }
    if (tid == 0) bsums[t * BATCH + b] = sred[0];
}

// ---------------------------------------------------------------------------
// Kernel 7: deterministic final loss reduction (no float atomics)
// ---------------------------------------------------------------------------
__global__ __launch_bounds__(256) void loss_reduce_kernel(const float* __restrict__ bsums,
                                                          float* __restrict__ total) {
    __shared__ float s[256];
    const int tid = threadIdx.x;
    float a = 0.0f;
    for (int i = tid; i < BATCH * T_POS; i += 256) a += bsums[i];
    s[tid] = a;
    __syncthreads();
    for (int st = 128; st > 0; st >>= 1) {
        if (tid < st) s[tid] += s[tid + st];
        __syncthreads();
    }
    if (tid == 0) total[0] = s[0] * (1.0f / (2.0f * T_POS * BATCH));
}

// ---------------------------------------------------------------------------
extern "C" void kernel_launch(void* const* d_in, const int* in_sizes, int n_in,
                              void* d_out, int out_size, void* d_ws, size_t ws_size,
                              hipStream_t stream) {
    (void)in_sizes; (void)n_in; (void)out_size; (void)ws_size;

    const float* x     = (const float*)d_in[0];
    const float* W_enc = (const float*)d_in[1];
    const float* b_enc = (const float*)d_in[2];
    const float* Wd0   = (const float*)d_in[3];
    const float* bd0   = (const float*)d_in[4];
    const float* Wd1   = (const float*)d_in[5];
    const float* bd1   = (const float*)d_in[6];
    const int*   perm  = (const int*)d_in[7];

    float* out   = (float*)d_out;
    float* total = out;                                    // [1]
    float* xhat  = out + 1;                                // [1024*4*768]
    float* z     = out + 1 + (size_t)BATCH * T_POS * D_IN; // [1024*16384], also pre scratch

    char* ws = (char*)d_ws;
    int*   idxs  = (int*)ws;                               // 1024*64 ints
    float* vals  = (float*)(ws + 1 * 65536 * 4);           // 1024*64 floats
    int*   inv   = (int*)(ws + 2 * 65536 * 4);             // 4*16384 ints
    float* bsums = (float*)(ws + 3 * 65536 * 4);           // 4096 floats

    invperm_kernel<<<(T_POS * D_SAE) / 256, 256, 0, stream>>>(perm, inv);

    enc_gemm_kernel<<<dim3(D_SAE / 256, BATCH / 64), 256, 0, stream>>>(
        x, W_enc, b_enc, z /* pre scratch */);

    topk_kernel<<<BATCH, 256, 0, stream>>>(z, idxs, vals);

    zero_z_kernel<<<4096, 256, 0, stream>>>(z);

    scatter_kernel<<<(BATCH * KTOP) / 256, 256, 0, stream>>>(idxs, vals, z);

    decode_kernel<<<dim3(BATCH, T_POS), 256, 0, stream>>>(
        x, idxs, vals, inv, Wd0, bd0, Wd1, bd1, xhat, bsums);

    loss_reduce_kernel<<<1, 256, 0, stream>>>(bsums, total);
}
